// DeepBayesianDKVMN_39857296507058
// MI455X (gfx1250) — matrix operand order repaired and checked by reading, not compiled, not run on hardware
//
#include <hip/hip_runtime.h>
#include <math.h>

// Problem constants
#define BB 1024
#define SS 512
#define MM 64
#define VV 128
#define EE 64
#define NQF 10000.0f

typedef __attribute__((ext_vector_type(16))) __bf16 v16bf;
typedef __attribute__((ext_vector_type(8)))  float  v8f;

// ---- workspace layout (bytes) ----
static constexpr size_t OFF_ATTN = 0;                               // S*B*M f32 = 128 MB  [S][B][M]
static constexpr size_t OFF_U    = OFF_ATTN + (size_t)SS*BB*MM*4;   // S*M*V f32 = 16 MB   [S][M][V]
static constexpr size_t OFF_C    = OFF_U    + (size_t)SS*MM*VV*4;   // S*M f32
static constexpr size_t OFF_E    = OFF_C    + (size_t)SS*MM*4;      // S*M f32
static constexpr size_t OFF_Z    = OFF_E    + (size_t)SS*MM*4;      // S*M f32
static constexpr size_t OFF_ADDV = OFF_Z    + (size_t)SS*MM*4;      // S*M f32
static constexpr size_t OFF_W    = OFF_ADDV + (size_t)SS*MM*4;      // E*M f32
static constexpr size_t OFF_BIAS = OFF_W    + (size_t)EE*MM*4;      // M f32
static constexpr size_t OFF_G    = OFF_BIAS + 256;                  // 2*V f32
static constexpr size_t OFF_G0   = OFF_G    + 1024;                 // V f32

__device__ __forceinline__ unsigned short f2bf(float f) {
  unsigned u = __builtin_bit_cast(unsigned, f);
  u += 0x7fffu + ((u >> 16) & 1u);                 // round-to-nearest-even
  return (unsigned short)(u >> 16);
}
__device__ __forceinline__ float softplusf(float x) {
  return (x > 0.f) ? (x + log1pf(expf(-x))) : log1pf(expf(x));
}

// ISA 16-bit A-matrix 16x32 fragment permutation: element (row r, k) -> (lane, half)
__device__ __forceinline__ int a_lane(int r, int k) { return r + 16 * ((k >> 3) & 1); }
__device__ __forceinline__ int a_half(int k)        { return (k & 7) + 8 * (k >> 4); }
// ISA 16-bit B-matrix 32x16 fragment permutation: element (k, col n) -> (lane, half)
__device__ __forceinline__ int b_lane(int n, int k) { return n + 16 * ((k >> 3) & 1); }
__device__ __forceinline__ int b_half(int k)        { return (k & 7) + 8 * (k >> 4); }

// ---------------- Kernel 0a: fold small matmuls ----------------
// W[e][m] = q2k_w @ key_embeds^T ; bias[m] = q2k_b @ key_embeds^T
// G[2][V] = qa_w @ qae_w ; g0[V] = qa_b @ qae_w + qae_b
__global__ __launch_bounds__(256) void prep0_kernel(
    const float* __restrict__ q2k_w, const float* __restrict__ q2k_b,
    const float* __restrict__ key_embeds,
    const float* __restrict__ qa_w, const float* __restrict__ qa_b,
    const float* __restrict__ qae_w, const float* __restrict__ qae_b,
    float* __restrict__ Wout, float* __restrict__ bias,
    float* __restrict__ G, float* __restrict__ g0) {
  int tid = threadIdx.x;
  for (int idx = tid; idx < EE * MM; idx += 256) {
    int e = idx >> 6, m = idx & 63;
    float s = 0.f;
    for (int k = 0; k < EE; ++k) s += q2k_w[e * EE + k] * key_embeds[m * EE + k];
    Wout[e * MM + m] = s;
  }
  if (tid < MM) {
    float s = 0.f;
    for (int k = 0; k < EE; ++k) s += q2k_b[k] * key_embeds[tid * EE + k];
    bias[tid] = s;
  }
  if (tid < VV) {
    float s0 = 0.f, s1 = 0.f, sb = 0.f;
    for (int e = 0; e < EE; ++e) {
      float w = qae_w[e * VV + tid];
      s0 += qa_w[e] * w;          // qa_w row 0
      s1 += qa_w[EE + e] * w;     // qa_w row 1
      sb += qa_b[e] * w;
    }
    G[tid] = s0; G[VV + tid] = s1; g0[tid] = sb + qae_b[tid];
  }
}

// ---------------- Kernel 0b: additive logit term addv[s][m] ----------------
__global__ __launch_bounds__(256) void prep1_kernel(
    const float* __restrict__ am, const float* __restrict__ alv,
    const float* __restrict__ bb, const float* __restrict__ bo,
    const float* __restrict__ an, const float* __restrict__ bn,
    const float* __restrict__ bias, float* __restrict__ addv) {
  int idx = blockIdx.x * 256 + threadIdx.x;
  if (idx >= SS * MM) return;
  int m = idx & 63;
  float alpha = expf(am[m] + an[idx] * expf(0.5f * alv[m]));
  // cum.mean over C-1=3 cols = (0 + sp0 + sp0+sp1)/3
  float cmm = (2.f * softplusf(bo[m * 3 + 0]) + softplusf(bo[m * 3 + 1])) * (1.f / 3.f);
  float bmn = bb[m] + bn[idx] * 0.1f + cmm;
  addv[idx] = 0.3f * alpha + 0.2f * expf(-0.5f * bmn * bmn) + bias[m];
}

// ---------------- Phase 1: attention via WMMA + softmax ----------------
// grid (8, 512): x = 128-batch chunk, y = s.  attn layout [S][B][M].
// LDS tiles staged directly in WMMA fragment order -> each fragment is a
// single 32-byte contiguous per-lane load (2x ds_load_b128).
__global__ __launch_bounds__(256) void attn_kernel(
    const int* __restrict__ q_ids, const float* __restrict__ q_table,
    const float* __restrict__ Wg, const float* __restrict__ addv,
    float* __restrict__ attn) {
  __shared__ v16bf qeF[8][2][32];          // A-frags: [wave(row-tile)][kc][lane]
  __shared__ v16bf WF[2][4][32];           // B-frags: [kc][mt][lane]
  __shared__ float lg[128][65];            // f32 logits
  __shared__ float av[64];
  __shared__ int qidl[128];
  unsigned short* qeFp = (unsigned short*)qeF;
  unsigned short* WFp  = (unsigned short*)WF;
  int tid = threadIdx.x;
  int s = blockIdx.y;
  int b0 = blockIdx.x * 128;

  if (tid < 64) av[tid] = addv[s * 64 + tid];
  if (tid < 128) qidl[tid] = q_ids[(size_t)(b0 + tid) * SS + s];
  for (int idx = tid; idx < EE * MM; idx += 256) {
    int e = idx >> 6, m = idx & 63;               // W[e][m]
    int kc = e >> 5, k = e & 31;
    int mt = m >> 4, n = m & 15;
    WFp[((kc * 4 + mt) * 32 + b_lane(n, k)) * 16 + b_half(k)] = f2bf(Wg[idx]);
  }
  __syncthreads();
  for (int idx = tid; idx < 128 * EE; idx += 256) {
    int r = idx >> 6, e = idx & 63;               // q_emb[r][e]
    int wv = r >> 4, rr = r & 15;
    int kc = e >> 5, k = e & 31;
    qeFp[((wv * 2 + kc) * 32 + a_lane(rr, k)) * 16 + a_half(k)] =
        f2bf(q_table[(size_t)qidl[r] * EE + e]);
  }
  __syncthreads();

  int lane = tid & 31, wave = tid >> 5;
  int r0 = wave * 16;
  int rlo = lane & 15;
  v8f acc[4] = {};
#pragma unroll
  for (int kc = 0; kc < 2; ++kc) {
    v16bf a = qeF[wave][kc][lane];                 // 2x ds_load_b128
#pragma unroll
    for (int mt = 0; mt < 4; ++mt) {
      v16bf b = WF[kc][mt][lane];                  // 2x ds_load_b128
      acc[mt] = __builtin_amdgcn_wmma_f32_16x16x32_bf16(
          false, a, false, b, (short)0, acc[mt], false, false);
    }
  }
  int rhi = (lane >= 16) ? 8 : 0;
#pragma unroll
  for (int mt = 0; mt < 4; ++mt)
#pragma unroll
    for (int i = 0; i < 8; ++i)
      lg[r0 + i + rhi][mt * 16 + rlo] = acc[mt][i] + av[mt * 16 + rlo];
  __syncthreads();

  if (tid < 128) {
    int r = tid;
    float mx = -1e30f;
    for (int m = 0; m < 64; ++m) mx = fmaxf(mx, lg[r][m]);
    float sum = 0.f;
    for (int m = 0; m < 64; ++m) sum += expf(lg[r][m] - mx);
    float inv = 1.f / sum;
    size_t base = ((size_t)s * BB + (b0 + r)) * MM;
    for (int m = 0; m < 64; ++m) attn[base + m] = expf(lg[r][m] - mx) * inv;
  }
}

// ---------------- Phase 2: per-step stats. grid 512 (one WG per t) ----------------
// colsum[m], s[m] -> c,e ;  U[m][v] = sum_b attn^2 * comb  via WMMA over B chunks.
__global__ __launch_bounds__(256) void stats_kernel(
    const int* __restrict__ q_ids, const int* __restrict__ responses,
    const float* __restrict__ attn, const float* __restrict__ Gg,
    const float* __restrict__ g0g, float* __restrict__ U,
    float* __restrict__ Carr, float* __restrict__ Earr) {
  __shared__ float pra[1024];
  __shared__ float Gl[2][128];
  __shared__ float g0l[128];
  __shared__ float part_cs[256], part_s[256];
  __shared__ v16bf aF[4][32];               // A-frags: [mt][lane]  (attn^2)
  __shared__ v16bf bF[8][32];               // B-frags: [vt][lane]  (comb)
  __shared__ float x0a[32], x1a[32], aea[32];
  unsigned short* aFp = (unsigned short*)aF;
  unsigned short* bFp = (unsigned short*)bF;
  int tid = threadIdx.x;
  int t = blockIdx.x;

  for (int b = tid; b < 1024; b += 256) {
    float rn = (float)responses[(size_t)b * SS + t] * (1.f / 3.f);
    pra[b] = 0.5f + fabsf(rn - 0.5f) * 2.f;
  }
  if (tid < 128) { Gl[0][tid] = Gg[tid]; Gl[1][tid] = Gg[128 + tid]; g0l[tid] = g0g[tid]; }
  __syncthreads();

  {  // colsum & s reductions (coalesced over m)
    int m = tid & 63, bg = tid >> 6;
    float acs = 0.f, as = 0.f;
    size_t base = (size_t)t * BB * MM;
    for (int b = bg; b < 1024; b += 4) {
      __builtin_prefetch(&attn[base + (size_t)(b + 16) * 64 + m], 0, 1);
      float a = attn[base + (size_t)b * 64 + m];
      acs += a; as += a * pra[b];
    }
    part_cs[tid] = acs; part_s[tid] = as;
  }
  __syncthreads();
  if (tid < 64) {
    float cs = part_cs[tid] + part_cs[tid + 64] + part_cs[tid + 128] + part_cs[tid + 192];
    float sv = part_s[tid] + part_s[tid + 64] + part_s[tid + 128] + part_s[tid + 192];
    float inv = 1.f / (cs + 1e-8f);
    float c = (sv * inv) * (cs * (1.f / 1024.f));   // agg_pr * w
    Carr[t * 64 + tid] = c;
    Earr[t * 64 + tid] = c * inv;                   // c / (cs+eps)
  }

  int lane = tid & 31, wave = tid >> 5;
  int mt = wave & 3, vt0 = (wave >> 2) * 4;
  int rlo = lane & 15;
  v8f acc[4] = {};
  for (int kc = 0; kc < 32; ++kc) {
    int b0 = kc * 32;
    __syncthreads();
    if (tid < 32) {
      int b = b0 + tid;
      float rn = (float)responses[(size_t)b * SS + t] * (1.f / 3.f);
      x0a[tid] = (float)q_ids[(size_t)b * SS + t] * (1.f / NQF);
      x1a[tid] = rn;
      float p = fminf(fmaxf(rn, 0.01f), 0.99f);
      aea[tid] = logf(p) - log1pf(-p);
    }
    {  // stage attn^2 tile in A-fragment order: rows m, K = local batch index kk
      size_t base = (size_t)t * BB * MM + (size_t)b0 * 64;
      for (int idx = tid; idx < 32 * 64; idx += 256) {
        int kk = idx >> 6, m = idx & 63;
        float a = attn[base + (size_t)kk * 64 + m];
        int mtx = m >> 4, r = m & 15;
        aFp[((mtx * 32) + a_lane(r, kk)) * 16 + a_half(kk)] = f2bf(a * a);
      }
    }
    __syncthreads();
    // stage comb tile in B-fragment order: K = kk, cols v
    for (int idx = tid; idx < 32 * 128; idx += 256) {
      int kk = idx >> 7, v = idx & 127;
      float val = 0.5f * (x0a[kk] * Gl[0][v] + x1a[kk] * Gl[1][v] + g0l[v] + aea[kk]);
      int vt = v >> 4, n = v & 15;
      bFp[((vt * 32) + b_lane(n, kk)) * 16 + b_half(kk)] = f2bf(val);
    }
    __syncthreads();

    v16bf a = aF[mt][lane];                        // 2x ds_load_b128
#pragma unroll
    for (int j = 0; j < 4; ++j) {
      v16bf b = bF[vt0 + j][lane];                 // 2x ds_load_b128
      acc[j] = __builtin_amdgcn_wmma_f32_16x16x32_bf16(
          false, a, false, b, (short)0, acc[j], false, false);
    }
  }
  int rhi = (lane >= 16) ? 8 : 0;
  size_t ub = (size_t)t * MM * VV;
#pragma unroll
  for (int j = 0; j < 4; ++j)
#pragma unroll
    for (int i = 0; i < 8; ++i) {
      int m = mt * 16 + i + rhi;
      int v = (vt0 + j) * 16 + rlo;
      U[ub + (size_t)m * VV + v] = acc[j][i];
    }
}

// ---------------- Phase 3: linearized scan. grid 64 (m) x 128 (v) ----------------
// Y_{t+1} = Y_t + e_t[m]*U_t[m,v]; prec_{t+1} = prec_t + c_t[m]
// z_t[m]  = sum_v (Y_t/prec_t)*pred_w[v]   (state BEFORE step-t update)
__global__ __launch_bounds__(128) void scan_kernel(
    const float* __restrict__ tm0, const float* __restrict__ tlv0,
    const float* __restrict__ pred_w, const float* __restrict__ U,
    const float* __restrict__ Carr, const float* __restrict__ Earr,
    float* __restrict__ Z) {
  int m = blockIdx.x;
  int v = threadIdx.x;
  int lane = v & 31, wid = v >> 5;
  __shared__ float wsum[4];
  float prec = expf(-tlv0[(size_t)m * VV + v]);
  float Y = prec * tm0[(size_t)m * VV + v];
  float pw = pred_w[v];
  for (int t = 0; t < SS; ++t) {
    __builtin_prefetch(&U[(size_t)(t + 2) * MM * VV + (size_t)m * VV + v], 0, 1);
    float u = (Y / prec) * pw;
    u += __shfl_xor(u, 16);
    u += __shfl_xor(u, 8);
    u += __shfl_xor(u, 4);
    u += __shfl_xor(u, 2);
    u += __shfl_xor(u, 1);
    if (lane == 0) wsum[wid] = u;
    __syncthreads();
    if (v == 0) Z[t * 64 + m] = wsum[0] + wsum[1] + wsum[2] + wsum[3];
    Y += Earr[t * 64 + m] * U[(size_t)t * MM * VV + (size_t)m * VV + v];
    prec += Carr[t * 64 + m];
    __syncthreads();
  }
}

// ---------------- Phase 4: preds. grid 512 (t). out[b][t] = attn_t[b,:].z_t + pb ----
__global__ __launch_bounds__(256) void pred_kernel(
    const float* __restrict__ attn, const float* __restrict__ Z,
    const float* __restrict__ pred_b, float* __restrict__ out) {
  __shared__ float zl[64];
  int tid = threadIdx.x, t = blockIdx.x;
  if (tid < 64) zl[tid] = Z[t * 64 + tid];
  __syncthreads();
  float pb = pred_b[0];
  int lane = tid & 31, wave = tid >> 5;
  size_t base = (size_t)t * BB * MM;
  for (int b = wave; b < BB; b += 8) {
    __builtin_prefetch(&attn[base + (size_t)(b + 8) * 64 + lane], 0, 1);
    float a0 = attn[base + (size_t)b * 64 + lane];
    float a1 = attn[base + (size_t)b * 64 + 32 + lane];
    float u = a0 * zl[lane] + a1 * zl[32 + lane];
    u += __shfl_xor(u, 16);
    u += __shfl_xor(u, 8);
    u += __shfl_xor(u, 4);
    u += __shfl_xor(u, 2);
    u += __shfl_xor(u, 1);
    if (lane == 0) out[(size_t)b * SS + t] = u + pb;
  }
}

extern "C" void kernel_launch(void* const* d_in, const int* in_sizes, int n_in,
                              void* d_out, int out_size, void* d_ws, size_t ws_size,
                              hipStream_t stream) {
  const int*   q_ids          = (const int*)d_in[0];
  const int*   responses      = (const int*)d_in[1];
  const float* q_table        = (const float*)d_in[2];
  const float* key_embeds     = (const float*)d_in[3];
  const float* alpha_mean     = (const float*)d_in[4];
  const float* alpha_log_var  = (const float*)d_in[5];
  const float* beta_base      = (const float*)d_in[6];
  const float* beta_offsets   = (const float*)d_in[7];
  const float* theta_mean0    = (const float*)d_in[8];
  const float* theta_log_var0 = (const float*)d_in[9];
  const float* q2k_w          = (const float*)d_in[10];
  const float* q2k_b          = (const float*)d_in[11];
  const float* qa_w           = (const float*)d_in[12];
  const float* qa_b           = (const float*)d_in[13];
  const float* qae_w          = (const float*)d_in[14];
  const float* qae_b          = (const float*)d_in[15];
  const float* pred_w         = (const float*)d_in[16];
  const float* pred_b         = (const float*)d_in[17];
  const float* alpha_noise    = (const float*)d_in[18];
  const float* beta_noise     = (const float*)d_in[19];

  char* ws = (char*)d_ws;
  float* attn = (float*)(ws + OFF_ATTN);
  float* U    = (float*)(ws + OFF_U);
  float* Carr = (float*)(ws + OFF_C);
  float* Earr = (float*)(ws + OFF_E);
  float* Z    = (float*)(ws + OFF_Z);
  float* addv = (float*)(ws + OFF_ADDV);
  float* W    = (float*)(ws + OFF_W);
  float* bias = (float*)(ws + OFF_BIAS);
  float* G    = (float*)(ws + OFF_G);
  float* g0   = (float*)(ws + OFF_G0);
  float* out  = (float*)d_out;

  prep0_kernel<<<1, 256, 0, stream>>>(q2k_w, q2k_b, key_embeds, qa_w, qa_b,
                                      qae_w, qae_b, W, bias, G, g0);
  prep1_kernel<<<(SS * MM + 255) / 256, 256, 0, stream>>>(
      alpha_mean, alpha_log_var, beta_base, beta_offsets,
      alpha_noise, beta_noise, bias, addv);
  attn_kernel<<<dim3(BB / 128, SS), 256, 0, stream>>>(q_ids, q_table, W, addv, attn);
  stats_kernel<<<SS, 256, 0, stream>>>(q_ids, responses, attn, G, g0, U, Carr, Earr);
  scan_kernel<<<MM, VV, 0, stream>>>(theta_mean0, theta_log_var0, pred_w, U, Carr, Earr, Z);
  pred_kernel<<<SS, 256, 0, stream>>>(attn, Z, pred_b, out);
}
